// ScannedRMT_21784074126080
// MI455X (gfx1250) — matrix operand order repaired and checked by reading, not compiled, not run on hardware
//
#include <hip/hip_runtime.h>

// ---------------- problem constants ----------------
#define Tt   256
#define Bb   256
#define Dd   512
#define Hh   8
#define HDim 64
#define Ll   4
#define MLPd 2048

// ---------------- launch config --------------------
#define NWG      128
#define TPB      256
#define WPB      (TPB/32)
#define NWAVES   (NWG*WPB)
#define NTHREADS (NWG*TPB)

typedef __attribute__((ext_vector_type(16))) __bf16 v16bf;
typedef __attribute__((ext_vector_type(8)))  __bf16 v8bf;
typedef __attribute__((ext_vector_type(8)))  float  v8f;

// ---------------- grid-wide barrier ----------------
__device__ __forceinline__ void gbar(unsigned* bar) {
    __syncthreads();
    if (threadIdx.x == 0) {
        __threadfence();
        unsigned g = __hip_atomic_load(&bar[1], __ATOMIC_RELAXED, __HIP_MEMORY_SCOPE_AGENT);
        unsigned a = __hip_atomic_fetch_add(&bar[0], 1u, __ATOMIC_ACQ_REL, __HIP_MEMORY_SCOPE_AGENT);
        if (a == (unsigned)(NWG - 1)) {
            __hip_atomic_store(&bar[0], 0u, __ATOMIC_RELAXED, __HIP_MEMORY_SCOPE_AGENT);
            __hip_atomic_fetch_add(&bar[1], 1u, __ATOMIC_RELEASE, __HIP_MEMORY_SCOPE_AGENT);
        } else {
            while (__hip_atomic_load(&bar[1], __ATOMIC_ACQUIRE, __HIP_MEMORY_SCOPE_AGENT) == g)
                __builtin_amdgcn_s_sleep(1);
        }
        __threadfence();
    }
    __syncthreads();
}

// ---------------- WMMA GEMM phase ------------------
// A: row-major bf16 [M x K], lda = K
// Bp: packed B fragments: tile (nt,kt) -> 32 lanes x 16 bf16 contiguous
// NB: N-tiles blocked per wave work unit (A-fragment reuse)
// MODE 0: outF[row*ldo + n]  = acc + bias
// MODE 1: outF[row*ldo + n] += acc + bias          (residual)
// MODE 2: outB[row*ldob + n] = bf16(gelu(acc+bias))
template <int MODE, int NB>
__device__ __forceinline__ void gemm_bf16(
    const __bf16* __restrict__ A, int lda,
    const __bf16* __restrict__ Bp,
    int M, int N, int K,
    const float* __restrict__ bias,
    float* __restrict__ outF, int ldo,
    __bf16* __restrict__ outB, int ldob,
    int wid, int lane)
{
    const int MT = M >> 4, NT = N >> 4, KT = K >> 5;
    const int units = MT * (NT / NB);
    const int half = lane >> 4;
    const int nn   = lane & 15;
    for (int u = wid; u < units; u += NWAVES) {
        const int mt  = u % MT;
        const int nt0 = (u / MT) * NB;
        const int m0  = mt << 4;
        const __bf16* __restrict__ arow = A + (size_t)(m0 + nn) * lda + half * 8;
        const __bf16* __restrict__ bp[NB];
        v8f acc[NB];
#pragma unroll
        for (int j = 0; j < NB; ++j) {
            bp[j] = Bp + ((size_t)(nt0 + j) * KT) * 512 + lane * 16;
            acc[j] = (v8f){0.f,0.f,0.f,0.f,0.f,0.f,0.f,0.f};
        }
        // prefetch next unit's weight fragments into L2/L1
        {
            const int un = u + NWAVES;
            if (un < units)
                __builtin_prefetch(Bp + ((size_t)((un / MT) * NB) * KT) * 512, 0, 1);
        }
        for (int kt = 0; kt < KT; ++kt) {
            const v8bf a0 = *(const v8bf*)(arow + kt * 32);
            const v8bf a1 = *(const v8bf*)(arow + kt * 32 + 16);
            v16bf a;
#pragma unroll
            for (int i = 0; i < 8; ++i) { a[i] = a0[i]; a[8 + i] = a1[i]; }
#pragma unroll
            for (int j = 0; j < NB; ++j) {
                const v16bf b = *(const v16bf*)(bp[j] + (size_t)kt * 512);
                acc[j] = __builtin_amdgcn_wmma_f32_16x16x32_bf16(
                             false, a, false, b, (short)0, acc[j], false, false);
            }
        }
        const int mrow = m0 + half * 8;
#pragma unroll
        for (int j = 0; j < NB; ++j) {
            const int outn = (nt0 + j) * 16 + nn;
            const float bv = bias ? bias[outn] : 0.f;
#pragma unroll
            for (int r = 0; r < 8; ++r) {
                float v = acc[j][r] + bv;
                const int row = mrow + r;
                if (MODE == 0) {
                    outF[(size_t)row * ldo + outn] = v;
                } else if (MODE == 1) {
                    outF[(size_t)row * ldo + outn] += v;
                } else {
                    float g = 0.5f * v * (1.f + tanhf(0.7978845608f * (v + 0.044715f * v * v * v)));
                    outB[(size_t)row * ldob + outn] = (__bf16)g;
                }
            }
        }
    }
}

// -------- mem_tok GEMM with fused done-mask + f32->bf16 convert --------
// A = memWs (f32, [Bb x Dd]), masked per row by dones[t*Bb + row].
// Output: seq even rows (row b -> seq[2b]),  M=256, N=512, K=512, no bias.
__device__ __forceinline__ void gemm_memtok(
    const float* __restrict__ memF, const unsigned char* __restrict__ dones, int t,
    const __bf16* __restrict__ Bp, float* __restrict__ seq,
    int wid, int lane)
{
    const int MT = Bb >> 4, NT = Dd >> 4, KT = Dd >> 5;
    const int half = lane >> 4;
    const int nn   = lane & 15;
    for (int u = wid; u < MT * NT; u += NWAVES) {
        const int mt = u % MT, nt = u / MT;
        const int m0 = mt << 4;
        const int row = m0 + nn;                 // batch index for this lane's A row
        const float msk = dones[t * Bb + row] ? 0.f : 1.f;
        const float* __restrict__ arow = memF + (size_t)row * Dd + half * 8;
        const __bf16* __restrict__ bptr = Bp + ((size_t)nt * KT) * 512 + lane * 16;
        v8f acc = {0.f,0.f,0.f,0.f,0.f,0.f,0.f,0.f};
        for (int kt = 0; kt < KT; ++kt) {
            const v8f x0 = *(const v8f*)(arow + kt * 32);
            const v8f x1 = *(const v8f*)(arow + kt * 32 + 16);
            v16bf a;
#pragma unroll
            for (int i = 0; i < 8; ++i) {
                a[i]     = (__bf16)(x0[i] * msk);
                a[8 + i] = (__bf16)(x1[i] * msk);
            }
            const v16bf b = *(const v16bf*)(bptr + (size_t)kt * 512);
            acc = __builtin_amdgcn_wmma_f32_16x16x32_bf16(
                      false, a, false, b, (short)0, acc, false, false);
        }
        const int outn = nt * 16 + nn;
        const int mrow = m0 + half * 8;
#pragma unroll
        for (int r = 0; r < 8; ++r)              // batch r -> seq row 2r  (ldo = 2*Dd)
            seq[(size_t)(mrow + r) * (2 * Dd) + outn] = acc[r];
    }
}

// ---------------- LayerNorm (wave per row) ---------
__device__ __forceinline__ void layernorm(
    const float* __restrict__ seq, const float* __restrict__ sc,
    const float* __restrict__ bs, __bf16* __restrict__ hbf,
    int wid, int lane)
{
    for (int row = wid; row < 2 * Bb; row += NWAVES) {
        const float* x = seq + (size_t)row * Dd;
        float vals[16], sum = 0.f, sq = 0.f;
#pragma unroll
        for (int i = 0; i < 16; ++i) {
            float v = x[lane + 32 * i];
            vals[i] = v; sum += v; sq += v * v;
        }
#pragma unroll
        for (int m = 16; m > 0; m >>= 1) {
            sum += __shfl_xor(sum, m, 32);
            sq  += __shfl_xor(sq,  m, 32);
        }
        const float mu = sum * (1.f / Dd);
        const float var = sq * (1.f / Dd) - mu * mu;
        const float rs = rsqrtf(var + 1e-6f);
#pragma unroll
        for (int i = 0; i < 16; ++i) {
            const int d = lane + 32 * i;
            hbf[(size_t)row * Dd + d] = (__bf16)((vals[i] - mu) * rs * sc[d] + bs[d]);
        }
    }
}

// ---------------- 2x2 attention (wave per (b,h)) ---
__device__ __forceinline__ void attention(
    const float* __restrict__ q, const float* __restrict__ k,
    const float* __restrict__ v, __bf16* __restrict__ obf,
    int wid, int lane)
{
    for (int u = wid; u < Bb * Hh; u += NWAVES) {
        const int b = u >> 3, h = u & 7;
        const size_t base = (size_t)(2 * b) * Dd + h * HDim;
        const int f0 = lane * 2, f1 = f0 + 1;
        float q00 = q[base + f0],      q01 = q[base + f1];
        float q10 = q[base + Dd + f0], q11 = q[base + Dd + f1];
        float k00 = k[base + f0],      k01 = k[base + f1];
        float k10 = k[base + Dd + f0], k11 = k[base + Dd + f1];
        float v00 = v[base + f0],      v01 = v[base + f1];
        float v10 = v[base + Dd + f0], v11 = v[base + Dd + f1];
        float s00 = q00 * k00 + q01 * k01;
        float s01 = q00 * k10 + q01 * k11;
        float s10 = q10 * k00 + q11 * k01;
        float s11 = q10 * k10 + q11 * k11;
#pragma unroll
        for (int m = 16; m > 0; m >>= 1) {
            s00 += __shfl_xor(s00, m, 32); s01 += __shfl_xor(s01, m, 32);
            s10 += __shfl_xor(s10, m, 32); s11 += __shfl_xor(s11, m, 32);
        }
        const float sc = 0.125f; // 1/sqrt(64)
        s00 *= sc; s01 *= sc; s10 *= sc; s11 *= sc;
        float m0 = fmaxf(s00, s01), m1 = fmaxf(s10, s11);
        float e00 = __expf(s00 - m0), e01 = __expf(s01 - m0);
        float e10 = __expf(s10 - m1), e11 = __expf(s11 - m1);
        float i0 = 1.f / (e00 + e01), i1 = 1.f / (e10 + e11);
        float a00 = e00 * i0, a01 = e01 * i0;
        float a10 = e10 * i1, a11 = e11 * i1;
        obf[base + f0]      = (__bf16)(a00 * v00 + a01 * v10);
        obf[base + f1]      = (__bf16)(a00 * v01 + a01 * v11);
        obf[base + Dd + f0] = (__bf16)(a10 * v00 + a11 * v10);
        obf[base + Dd + f1] = (__bf16)(a10 * v01 + a11 * v11);
    }
}

// ---------------- weight packing -------------------
// packed index = ((nt*KT + kt)*32 + lane)*16 + e ;
// k = kt*32 + (e>>3)*16 + (lane>>4)*8 + (e&7) ; n = nt*16 + (lane&15)
__global__ void pack_w(const float* __restrict__ src, __bf16* __restrict__ dst, int K, int N) {
    const int total = K * N;
    const int KT = K >> 5;
    for (int idx = blockIdx.x * blockDim.x + threadIdx.x; idx < total;
         idx += gridDim.x * blockDim.x) {
        const int tile = idx >> 9;
        const int within = idx & 511;
        const int lane = within >> 4;
        const int e = within & 15;
        const int nt = tile / KT;
        const int kt = tile - nt * KT;
        const int k = kt * 32 + (e >> 3) * 16 + (lane >> 4) * 8 + (e & 7);
        const int n = nt * 16 + (lane & 15);
        dst[idx] = (__bf16)src[(size_t)k * N + n];
    }
}

__global__ void init_ws(unsigned* bar, const float* __restrict__ memIn, float* __restrict__ memWs) {
    const int i = blockIdx.x * blockDim.x + threadIdx.x;
    if (i < 8) bar[i] = 0u;
    for (int j = i; j < Bb * Dd; j += gridDim.x * blockDim.x) memWs[j] = memIn[j];
}

// ---------------- persistent scan kernel -----------
__global__ void __launch_bounds__(TPB) rmt_persistent(
    const float* __restrict__ obs, const unsigned char* __restrict__ dones,
    const float* __restrict__ ln1_s, const float* __restrict__ ln1_b,
    const float* __restrict__ bq, const float* __restrict__ bk,
    const float* __restrict__ bv, const float* __restrict__ bo,
    const float* __restrict__ ln2_s, const float* __restrict__ ln2_b,
    const float* __restrict__ b1, const float* __restrict__ b2,
    const __bf16* __restrict__ Wbf, float* __restrict__ out,
    unsigned* bar, float* memWs, float* seq, __bf16* hbf,
    float* qB, float* kB, float* vB, __bf16* obf, __bf16* gbf)
{
    const int lane = threadIdx.x & 31;
    const int wid  = blockIdx.x * WPB + (threadIdx.x >> 5);
    const int gtid = blockIdx.x * TPB + threadIdx.x;
    const size_t lstride = (size_t)4 * Dd * Dd + (size_t)2 * Dd * MLPd;

    // prologue: stage obs[0] into odd seq rows
    for (int i = gtid; i < Bb * Dd; i += NTHREADS) {
        const int b = i >> 9, d = i & 511;
        seq[(size_t)(2 * b + 1) * Dd + d] = obs[i];
    }
    gbar(bar);

    for (int t = 0; t < Tt; ++t) {
        // mem_tok = mask(mem) @ mem_kernel -> seq even rows (mask+convert fused)
        gemm_memtok(memWs, dones, t, Wbf, seq, wid, lane);
        gbar(bar);

        for (int l = 0; l < Ll; ++l) {
            const __bf16* wl = Wbf + (size_t)Dd * Dd + (size_t)l * lstride;
            const __bf16* Wq = wl;
            const __bf16* Wk = wl + (size_t)Dd * Dd;
            const __bf16* Wv = wl + (size_t)2 * Dd * Dd;
            const __bf16* Wo = wl + (size_t)3 * Dd * Dd;
            const __bf16* W1 = wl + (size_t)4 * Dd * Dd;
            const __bf16* W2 = wl + (size_t)4 * Dd * Dd + (size_t)Dd * MLPd;

            layernorm(seq, ln1_s + l * Dd, ln1_b + l * Dd, hbf, wid, lane);
            gbar(bar);
            gemm_bf16<0,2>(hbf, Dd, Wq, 2 * Bb, Dd, Dd, bq + l * Dd, qB, Dd, nullptr, 0, wid, lane);
            gemm_bf16<0,2>(hbf, Dd, Wk, 2 * Bb, Dd, Dd, bk + l * Dd, kB, Dd, nullptr, 0, wid, lane);
            gemm_bf16<0,2>(hbf, Dd, Wv, 2 * Bb, Dd, Dd, bv + l * Dd, vB, Dd, nullptr, 0, wid, lane);
            gbar(bar);
            attention(qB, kB, vB, obf, wid, lane);
            gbar(bar);
            gemm_bf16<1,1>(obf, Dd, Wo, 2 * Bb, Dd, Dd, bo + l * Dd, seq, Dd, nullptr, 0, wid, lane);
            gbar(bar);
            layernorm(seq, ln2_s + l * Dd, ln2_b + l * Dd, hbf, wid, lane);
            gbar(bar);
            gemm_bf16<2,2>(hbf, Dd, W1, 2 * Bb, MLPd, Dd, b1 + l * MLPd, nullptr, 0, gbf, MLPd, wid, lane);
            gbar(bar);
            gemm_bf16<1,1>(gbf, MLPd, W2, 2 * Bb, Dd, MLPd, b2 + l * Dd, seq, Dd, nullptr, 0, wid, lane);
            gbar(bar);
        }
        // fused: emit output token, carry memory, stage next obs token
        for (int i = gtid; i < Bb * Dd; i += NTHREADS) {
            const int b = i >> 9, d = i & 511;
            const size_t sidx = (size_t)(2 * b + 1) * Dd + d;
            const float v = seq[sidx];
            out[(size_t)t * Bb * Dd + i] = v;
            memWs[i] = v;
            if (t + 1 < Tt) seq[sidx] = obs[(size_t)(t + 1) * Bb * Dd + i];
        }
        gbar(bar);
    }
}

// ---------------- host entry -----------------------
extern "C" void kernel_launch(void* const* d_in, const int* in_sizes, int n_in,
                              void* d_out, int out_size, void* d_ws, size_t ws_size,
                              hipStream_t stream) {
    const float*         obs   = (const float*)d_in[0];
    const unsigned char* dones = (const unsigned char*)d_in[1];
    const float*         memIn = (const float*)d_in[2];
    const float*         memK  = (const float*)d_in[3];
    const float* ln1_s = (const float*)d_in[4];
    const float* ln1_b = (const float*)d_in[5];
    const float* wq = (const float*)d_in[6];
    const float* bq = (const float*)d_in[7];
    const float* wk = (const float*)d_in[8];
    const float* bk = (const float*)d_in[9];
    const float* wv = (const float*)d_in[10];
    const float* bv = (const float*)d_in[11];
    const float* wo = (const float*)d_in[12];
    const float* bo = (const float*)d_in[13];
    const float* ln2_s = (const float*)d_in[14];
    const float* ln2_b = (const float*)d_in[15];
    const float* w1 = (const float*)d_in[16];
    const float* b1 = (const float*)d_in[17];
    const float* w2 = (const float*)d_in[18];
    const float* b2 = (const float*)d_in[19];
    float* out = (float*)d_out;

    char* ws = (char*)d_ws;
    size_t off = 0;
    auto carve = [&](size_t bytes) -> char* {
        char* p = ws + off;
        off += (bytes + 255) & ~(size_t)255;
        return p;
    };
    unsigned* bar  = (unsigned*)carve(256);
    float*   memWs = (float*)carve((size_t)Bb * Dd * 4);
    float*   seq   = (float*)carve((size_t)2 * Bb * Dd * 4);
    __bf16*  hbf   = (__bf16*)carve((size_t)2 * Bb * Dd * 2);
    float*   qB    = (float*)carve((size_t)2 * Bb * Dd * 4);
    float*   kB    = (float*)carve((size_t)2 * Bb * Dd * 4);
    float*   vB    = (float*)carve((size_t)2 * Bb * Dd * 4);
    __bf16*  obf   = (__bf16*)carve((size_t)2 * Bb * Dd * 2);
    __bf16*  gbf   = (__bf16*)carve((size_t)2 * Bb * MLPd * 2);
    const size_t lstride = (size_t)4 * Dd * Dd + (size_t)2 * Dd * MLPd;
    __bf16*  Wbf   = (__bf16*)carve(((size_t)Dd * Dd + (size_t)Ll * lstride) * 2);

    init_ws<<<256, 256, 0, stream>>>(bar, memIn, memWs);
    pack_w<<<512, 256, 0, stream>>>(memK, Wbf, Dd, Dd);
    for (int l = 0; l < Ll; ++l) {
        __bf16* wl = Wbf + (size_t)Dd * Dd + (size_t)l * lstride;
        pack_w<<<512,  256, 0, stream>>>(wq + (size_t)l * Dd * Dd, wl,                       Dd,   Dd);
        pack_w<<<512,  256, 0, stream>>>(wk + (size_t)l * Dd * Dd, wl + (size_t)Dd * Dd,     Dd,   Dd);
        pack_w<<<512,  256, 0, stream>>>(wv + (size_t)l * Dd * Dd, wl + (size_t)2 * Dd * Dd, Dd,   Dd);
        pack_w<<<512,  256, 0, stream>>>(wo + (size_t)l * Dd * Dd, wl + (size_t)3 * Dd * Dd, Dd,   Dd);
        pack_w<<<2048, 256, 0, stream>>>(w1 + (size_t)l * Dd * MLPd, wl + (size_t)4 * Dd * Dd, Dd, MLPd);
        pack_w<<<2048, 256, 0, stream>>>(w2 + (size_t)l * Dd * MLPd,
                                         wl + (size_t)4 * Dd * Dd + (size_t)Dd * MLPd, MLPd, Dd);
    }
    rmt_persistent<<<NWG, TPB, 0, stream>>>(
        obs, dones, ln1_s, ln1_b, bq, bk, bv, bo, ln2_s, ln2_b, b1, b2,
        Wbf, out, bar, memWs, seq, hbf, qB, kB, vB, obf, gbf);
}
